// BailingMoeV25MLAAttention_32633161515518
// MI455X (gfx1250) — compile-verified, matrix operand order checked by reading, not run on hardware
//
#include <hip/hip_runtime.h>
#include <hip/hip_bf16.h>
#include <math.h>

typedef __attribute__((ext_vector_type(16))) __bf16 v16bf;
typedef __attribute__((ext_vector_type(8)))  __bf16 v8bf;
typedef __attribute__((ext_vector_type(4)))  __bf16 v4bf;
typedef __attribute__((ext_vector_type(8)))  float  v8f;
typedef __attribute__((ext_vector_type(4)))  int    v4i;

#define AS1 __attribute__((address_space(1)))
#define AS3 __attribute__((address_space(3)))

#ifndef __has_builtin
#define __has_builtin(x) 0
#endif
#if __has_builtin(__builtin_amdgcn_global_load_async_to_lds_b128)
#define USE_ASYNC 1
#else
#define USE_ASYNC 0
#endif

// 16-byte global -> LDS copy. Async DMA path (ASYNCcnt) when available.
__device__ __forceinline__ void cp16(void* lds, const void* g) {
#if USE_ASYNC
  __builtin_amdgcn_global_load_async_to_lds_b128(
      (AS1 v4i*)const_cast<void*>(g), (AS3 v4i*)lds, 0, 0);
#else
  *(uint4*)lds = *(const uint4*)g;
#endif
}

__device__ __forceinline__ void async_wait() {
#if USE_ASYNC
  asm volatile("s_wait_asynccnt 0x0" ::: "memory");
#endif
}

// ---------------------------------------------------------------------------
// WMMA bf16 fragment loader from a row-major LDS tile (ISA 7.12.2 layout).
// Lane's 16 values = two contiguous 16B runs of row (lane&15):
// [kh, kh+8) and [16+kh, 16+kh+8), kh = 8*(lane>>4). Two ds_load_b128.
// ---------------------------------------------------------------------------
__device__ __forceinline__ v16bf load_frag(const __bf16* base, int ld) {
  const int lane = threadIdx.x & 31;
  const int row  = lane & 15;
  const int kh   = (lane >> 4) << 3;
  const __bf16* p = base + row * ld + kh;
  v8bf lo = *(const v8bf*)(p);
  v8bf hi = *(const v8bf*)(p + 16);
  return __builtin_shufflevector(lo, hi, 0, 1, 2, 3, 4, 5, 6, 7,
                                 8, 9, 10, 11, 12, 13, 14, 15);
}

__device__ __forceinline__ v8f wmma_bf16(v16bf a, v16bf b, v8f c) {
  return __builtin_amdgcn_wmma_f32_16x16x32_bf16(false, a, false, b,
                                                 (short)0, c, false, false);
}

// ---------------------------------------------------------------------------
// fp32 -> bf16 bulk convert (grid-stride, float4 reads, b64 writes). n%4==0.
// ---------------------------------------------------------------------------
__global__ __launch_bounds__(256) void f32_to_bf16_kernel(
    const float* __restrict__ src, __bf16* __restrict__ dst, long n) {
  long stride = (long)gridDim.x * 256 * 4;
  for (long i = ((long)blockIdx.x * 256 + threadIdx.x) * 4; i < n; i += stride) {
    float4 v = *(const float4*)(src + i);
    v4bf o = {(__bf16)v.x, (__bf16)v.y, (__bf16)v.z, (__bf16)v.w};
    *(v4bf*)(dst + i) = o;
  }
}

// ---------------------------------------------------------------------------
// GEMM: C[M,N](f32) = A[M,K](bf16) @ B[K,N](bf16), bf16 WMMA, f32 acc.
// M%128==0, N%64==0, K%32==0. 128 threads / 4 waves, 128x64 block tile;
// wave w owns rows [w*32, w*32+32) (2 A-frags x 4 B-frags = 8 WMMA/K-step).
// Double-buffered K loop: next tile's async A-copies + B register loads are
// issued before computing the current tile; waits drain after the WMMAs.
// ---------------------------------------------------------------------------
__global__ __launch_bounds__(128) void gemm_bf16_kernel(
    const __bf16* __restrict__ A, const __bf16* __restrict__ B,
    float* __restrict__ C, int M, int N, int K) {
  const int n0   = blockIdx.x * 64;
  const int m0   = blockIdx.y * 128;
  const int tid  = threadIdx.x;
  const int w    = tid >> 5;
  const int lane = tid & 31;
  const int col  = lane & 15;
  const int hi8  = (lane >> 4) << 3;

  __shared__ alignas(16) __bf16 As[2][128][32];  // 16 KB
  __shared__ alignas(16) __bf16 BsT[2][64][40];  // 10 KB (transposed, padded)

  // Per-thread staging coordinates (hoisted; pointers bumped per K tile).
  const int ar = tid >> 2, ac = (tid & 3) << 3;   // A: rows ar+32j, j=0..3
  const int br = tid >> 4, bc = (tid & 15) << 2;  // B: rows br+8j, cols bc..bc+3
  const __bf16* aptr = A + (size_t)(m0 + ar) * K + ac;
  const __bf16* bptr = B + (size_t)br * N + n0 + bc;
  const size_t aRow32 = (size_t)32 * K;
  const size_t bRow8  = (size_t)8 * N;

  // Prologue: stage K-tile 0 into buffer 0.
#pragma unroll
  for (int j = 0; j < 4; ++j) cp16(&As[0][ar + 32 * j][ac], aptr + j * aRow32);
  {
    v4bf t0 = *(const v4bf*)(bptr);
    v4bf t1 = *(const v4bf*)(bptr + bRow8);
    v4bf t2 = *(const v4bf*)(bptr + 2 * bRow8);
    v4bf t3 = *(const v4bf*)(bptr + 3 * bRow8);
#pragma unroll
    for (int i = 0; i < 4; ++i) {
      BsT[0][bc + i][br]      = t0[i];
      BsT[0][bc + i][br + 8]  = t1[i];
      BsT[0][bc + i][br + 16] = t2[i];
      BsT[0][bc + i][br + 24] = t3[i];
    }
  }
  async_wait();
  __syncthreads();

  v8f acc[8] = {}; // [0..3]: rows w*32+0..15; [4..7]: rows w*32+16..31
  int bb = 0;
  for (int k0 = 0;;) {
    const bool more = (k0 + 32) < K;
    v4bf t0, t1, t2, t3;
    if (more) { // issue next tile's loads before computing current
      aptr += 32;
      bptr += (size_t)32 * N;
#pragma unroll
      for (int j = 0; j < 4; ++j)
        cp16(&As[bb ^ 1][ar + 32 * j][ac], aptr + j * aRow32);
      t0 = *(const v4bf*)(bptr);
      t1 = *(const v4bf*)(bptr + bRow8);
      t2 = *(const v4bf*)(bptr + 2 * bRow8);
      t3 = *(const v4bf*)(bptr + 3 * bRow8);
    }

    v16bf af0 = load_frag(&As[bb][w * 32][0], 32);
    v16bf af1 = load_frag(&As[bb][w * 32 + 16][0], 32);
#pragma unroll
    for (int nt = 0; nt < 4; ++nt) {
      v16bf bf = load_frag(&BsT[bb][nt * 16][0], 40);
      acc[nt]     = wmma_bf16(af0, bf, acc[nt]);
      acc[nt + 4] = wmma_bf16(af1, bf, acc[nt + 4]);
    }
    if (!more) break;

    const int nb = bb ^ 1;
#pragma unroll
    for (int i = 0; i < 4; ++i) {
      BsT[nb][bc + i][br]      = t0[i];
      BsT[nb][bc + i][br + 8]  = t1[i];
      BsT[nb][bc + i][br + 16] = t2[i];
      BsT[nb][bc + i][br + 24] = t3[i];
    }
    async_wait();
    __syncthreads();
    bb = nb;
    k0 += 32;
  }

#pragma unroll
  for (int h2 = 0; h2 < 2; ++h2)
#pragma unroll
    for (int nt = 0; nt < 4; ++nt)
#pragma unroll
      for (int r = 0; r < 8; ++r)
        C[(size_t)(m0 + w * 32 + h2 * 16 + r + hi8) * N + n0 + nt * 16 + col] =
            acc[h2 * 4 + nt][r];
}

// ---------------------------------------------------------------------------
// RMSNorm: one block per token, float4 I/O, bf16 output. D%4==0.
// ---------------------------------------------------------------------------
__global__ __launch_bounds__(256) void rmsnorm_kernel(
    const float* __restrict__ in, int ld, int off, const float* __restrict__ w,
    __bf16* __restrict__ out, int D) {
  const int t = blockIdx.x;
  const float* x = in + (size_t)t * ld + off;
  __shared__ float red[256];
  float s = 0.f;
  for (int d = threadIdx.x * 4; d < D; d += 1024) {
    float4 v = *(const float4*)(x + d);
    s += v.x * v.x + v.y * v.y + v.z * v.z + v.w * v.w;
  }
  red[threadIdx.x] = s;
  __syncthreads();
  for (int st = 128; st > 0; st >>= 1) {
    if (threadIdx.x < st) red[threadIdx.x] += red[threadIdx.x + st];
    __syncthreads();
  }
  const float inv = rsqrtf(red[0] / (float)D + 1e-6f);
  for (int d = threadIdx.x * 4; d < D; d += 1024) {
    float4 v = *(const float4*)(x + d);
    float4 ww = *(const float4*)(w + d);
    v4bf o = {(__bf16)(v.x * inv * ww.x), (__bf16)(v.y * inv * ww.y),
              (__bf16)(v.z * inv * ww.z), (__bf16)(v.w * inv * ww.w)};
    *(v4bf*)(out + (size_t)t * D + d) = o;
  }
}

// ---------------------------------------------------------------------------
// RoPE (16 interleaved pairs over first 32 of 64 pe dims, rest passthrough).
// ---------------------------------------------------------------------------
__global__ void rope_k_kernel(const float* __restrict__ qkv,
                              const int* __restrict__ pos,
                              float* __restrict__ kpe, int ld, int off) {
  const int t = blockIdx.x;
  const float* x = qkv + (size_t)t * ld + off;
  float* y = kpe + (size_t)t * 64;
  const int tid = threadIdx.x;
  const float p = (float)pos[t];
  if (tid < 16) {
    float inv = powf(600000.0f, -(float)(2 * tid) / 32.0f);
    float fr = p * inv;
    float c = cosf(fr), s = sinf(fr);
    float x1 = x[2 * tid], x2 = x[2 * tid + 1];
    y[2 * tid]     = x1 * c - x2 * s;
    y[2 * tid + 1] = x2 * c + x1 * s;
  } else if (tid >= 32 && tid < 64) {
    y[tid] = x[tid];
  }
}

__global__ void rope_q_kernel(float* __restrict__ q,
                              const int* __restrict__ pos) {
  const int t = blockIdx.x, h = blockIdx.y;
  float* x = q + (size_t)t * 3072 + h * 192 + 128;
  const int tid = threadIdx.x;
  if (tid < 16) {
    float p = (float)pos[t];
    float inv = powf(600000.0f, -(float)(2 * tid) / 32.0f);
    float fr = p * inv;
    float c = cosf(fr), s = sinf(fr);
    float x1 = x[2 * tid], x2 = x[2 * tid + 1];
    x[2 * tid]     = x1 * c - x2 * s;
    x[2 * tid + 1] = x2 * c + x1 * s;
  }
}

// ---------------------------------------------------------------------------
// Build bf16 attention caches once:
//   qc[T,16,192] = q * 192^-0.5 (roped), kc[T,16,192] = {k_nope, kpe bcast},
//   vc[T,16,128] = v
// ---------------------------------------------------------------------------
__global__ __launch_bounds__(256) void build_caches_kernel(
    const float* __restrict__ qb, const float* __restrict__ kvb,
    const float* __restrict__ kpe, __bf16* __restrict__ qc,
    __bf16* __restrict__ kc, __bf16* __restrict__ vc) {
  const int t = blockIdx.x;
  const float kScale = 0.07216878364870322f; // 192^-0.5
  for (int i = threadIdx.x; i < 16 * 192; i += 256) {
    int h = i / 192, d = i % 192;
    qc[(size_t)t * 3072 + i] = (__bf16)(qb[(size_t)t * 3072 + i] * kScale);
    float kvv = (d < 128) ? kvb[(size_t)t * 4096 + h * 256 + d]
                          : kpe[(size_t)t * 64 + d - 128];
    kc[(size_t)t * 3072 + i] = (__bf16)kvv;
  }
  for (int i = threadIdx.x; i < 16 * 128; i += 256) {
    int h = i >> 7, d = i & 127;
    vc[(size_t)t * 2048 + i] = (__bf16)kvb[(size_t)t * 4096 + h * 256 + 128 + d];
  }
}

// ---------------------------------------------------------------------------
// Flash-style causal MLA attention over bf16 caches.
// Block = (64-query tile, head), 128 threads / 4 waves. Key tiles of 32.
// Output ob[T,16,128] in bf16 (A operand of the final GEMM).
// ---------------------------------------------------------------------------
__global__ __launch_bounds__(128) void mla_attn_kernel(
    const __bf16* __restrict__ qc, const __bf16* __restrict__ kc,
    const __bf16* __restrict__ vc, __bf16* __restrict__ ob, int T) {
  const int qt   = blockIdx.x;
  const int h    = blockIdx.y;
  const int tid  = threadIdx.x;
  const int w    = tid >> 5;
  const int lane = tid & 31;
  const int col  = lane & 15;
  const int hi8  = (lane >> 4) << 3;

  __shared__ alignas(16) __bf16 Qs[64][192];   // 24 KB
  __shared__ alignas(16) __bf16 Ks[32][192];   // 12 KB
  __shared__ alignas(16) __bf16 VsT[128][40];  // 10 KB (transposed, padded)
  __shared__ alignas(16) float  Ss[4][16][32]; //  8 KB
  __shared__ alignas(16) __bf16 Ps[4][16][32]; //  4 KB
  __shared__ float Mrow[4][16], Lrow[4][16], Arow[4][16];

  // Stage Q tile: 64 rows x 192 bf16 = 1536 x 16B chunks; 12 per thread
#pragma unroll
  for (int j = 0; j < 12; ++j) {
    int idx = tid + j * 128;
    int r = idx / 24, c8 = (idx % 24) << 3;
    cp16(&Qs[r][c8], &qc[(size_t)(qt * 64 + r) * 3072 + h * 192 + c8]);
  }
  if (tid < 64) {
    Mrow[tid >> 4][tid & 15] = -3.0e38f;
    Lrow[tid >> 4][tid & 15] = 0.0f;
  }

  // Hoisted per-thread staging pointers for K/V (bumped per key tile).
  const __bf16* kp[6];
  __bf16* kl[6];
#pragma unroll
  for (int j = 0; j < 6; ++j) {
    int idx = tid + j * 128;
    int r = idx / 24, c8 = (idx % 24) << 3;
    kp[j] = kc + (size_t)r * 3072 + h * 192 + c8;
    kl[j] = &Ks[r][c8];
  }
  const int vr = tid >> 5, vc4 = (tid & 31) << 2; // V: rows vr+8j, cols vc4..+3
  const __bf16* vp = vc + (size_t)vr * 2048 + h * 128 + vc4;

  async_wait();
  __syncthreads();

  // Loop-invariant Q fragments: 6 K-steps of 32 over d=192
  v16bf aq[6];
#pragma unroll
  for (int kk = 0; kk < 6; ++kk) aq[kk] = load_frag(&Qs[w * 16][kk * 32], 192);

  v8f acc[8] = {}; // 16 rows x 128 v-dims

  const int nkt = 2 * qt + 2;
  for (int kt = 0; kt < nkt; ++kt) {
    const int k0 = kt * 32;
    __syncthreads(); // Ks/VsT reused
    // K tile: async b128 copies
#pragma unroll
    for (int j = 0; j < 6; ++j) cp16(kl[j], kp[j]);
    // V tile transposed: batch 8 global v4 loads, then LDS stores
    v4bf vt[8];
#pragma unroll
    for (int j = 0; j < 8; ++j)
      vt[j] = *(const v4bf*)(vp + (size_t)(k0 + j * 8) * 2048);
#pragma unroll
    for (int j = 0; j < 8; ++j)
#pragma unroll
      for (int i = 0; i < 4; ++i) VsT[vc4 + i][vr + j * 8] = vt[j][i];
#pragma unroll
    for (int j = 0; j < 6; ++j) kp[j] += (size_t)32 * 3072;
    async_wait();
    __syncthreads();

    // S = Q @ K^T (16 rows x 32 keys): B operand is row-major Ks (== K^T cols)
    v8f sa[2] = {};
#pragma unroll
    for (int nt = 0; nt < 2; ++nt)
#pragma unroll
      for (int kk = 0; kk < 6; ++kk) {
        v16bf bf = load_frag(&Ks[nt * 16][kk * 32], 192);
        sa[nt] = wmma_bf16(aq[kk], bf, sa[nt]);
      }

#pragma unroll
    for (int nt = 0; nt < 2; ++nt)
#pragma unroll
      for (int r = 0; r < 8; ++r) Ss[w][r + hi8][nt * 16 + col] = sa[nt][r];
    __syncthreads();

    // Online softmax: one lane per row
    if (lane < 16) {
      const int r = lane;
      const int qidx = qt * 64 + w * 16 + r;
      float m = Mrow[w][r], l = Lrow[w][r];
      float sc[32];
      float tmax = -3.0e38f;
#pragma unroll
      for (int c4 = 0; c4 < 8; ++c4) {
        float4 sv = *(const float4*)&Ss[w][r][c4 * 4];
        sc[c4 * 4 + 0] = (k0 + c4 * 4 + 0 <= qidx) ? sv.x : -3.0e38f;
        sc[c4 * 4 + 1] = (k0 + c4 * 4 + 1 <= qidx) ? sv.y : -3.0e38f;
        sc[c4 * 4 + 2] = (k0 + c4 * 4 + 2 <= qidx) ? sv.z : -3.0e38f;
        sc[c4 * 4 + 3] = (k0 + c4 * 4 + 3 <= qidx) ? sv.w : -3.0e38f;
      }
#pragma unroll
      for (int c = 0; c < 32; ++c) tmax = fmaxf(tmax, sc[c]);
      const float nm = fmaxf(m, tmax);
      const float alpha = __expf(m - nm);
      float sum = 0.f;
#pragma unroll
      for (int c = 0; c < 32; ++c) {
        float p = (sc[c] > -1.0e38f) ? __expf(sc[c] - nm) : 0.0f;
        Ps[w][r][c] = (__bf16)p;
        sum += p;
      }
      Mrow[w][r] = nm;
      Lrow[w][r] = l * alpha + sum;
      Arow[w][r] = alpha;
    }
    __syncthreads();

    // acc = acc * alpha + P @ V
    v16bf pf = load_frag(&Ps[w][0][0], 32);
#pragma unroll
    for (int nt = 0; nt < 8; ++nt) {
#pragma unroll
      for (int r = 0; r < 8; ++r) acc[nt][r] *= Arow[w][r + hi8];
      v16bf vf = load_frag(&VsT[nt * 16][0], 40);
      acc[nt] = wmma_bf16(pf, vf, acc[nt]);
    }
  }

  __syncthreads();
#pragma unroll
  for (int nt = 0; nt < 8; ++nt)
#pragma unroll
    for (int r = 0; r < 8; ++r) {
      const float linv = 1.0f / Lrow[w][r + hi8];
      ob[(size_t)(qt * 64 + w * 16 + r + hi8) * 2048 + h * 128 + nt * 16 + col] =
          (__bf16)(acc[nt][r] * linv);
    }
}

// ---------------------------------------------------------------------------
// Host-side orchestration.
// ---------------------------------------------------------------------------
extern "C" void kernel_launch(void* const* d_in, const int* in_sizes, int n_in,
                              void* d_out, int out_size, void* d_ws,
                              size_t ws_size, hipStream_t stream) {
  const float* hidden      = (const float*)d_in[0];
  const int*   positions   = (const int*)d_in[1];
  const float* w_qkv_a     = (const float*)d_in[2];
  const float* q_a_norm_w  = (const float*)d_in[3];
  const float* w_q_b       = (const float*)d_in[4];
  const float* kv_a_norm_w = (const float*)d_in[5];
  const float* w_kv_b      = (const float*)d_in[6];
  const float* w_o         = (const float*)d_in[7];
  float* out = (float*)d_out;

  const int T = in_sizes[0] / 2048; // HIDDEN = 2048

  // fp32 workspace segment
  float* wsf = (float*)d_ws;
  float* qkv = wsf;                         // [T, 2112]
  float* qb  = qkv + (size_t)T * 2112;      // [T, 3072]
  float* kvb = qb + (size_t)T * 3072;       // [T, 4096]
  float* kpe = kvb + (size_t)T * 4096;      // [T, 64]
  // bf16 workspace segment
  __bf16* wsb      = (__bf16*)(kpe + (size_t)T * 64);
  __bf16* hidden_b = wsb;                          // [T, 2048]
  __bf16* wqkv_b   = hidden_b + (size_t)T * 2048;  // [2048, 2112]
  __bf16* wqb_b    = wqkv_b + (size_t)2048 * 2112; // [1536, 3072]
  __bf16* wkvb_b   = wqb_b + (size_t)1536 * 3072;  // [512, 4096]
  __bf16* wo_b     = wkvb_b + (size_t)512 * 4096;  // [2048, 2048]
  __bf16* qn       = wo_b + (size_t)2048 * 2048;   // [T, 1536]
  __bf16* kvn      = qn + (size_t)T * 1536;        // [T, 512]
  __bf16* qcache   = kvn + (size_t)T * 512;        // [T, 16, 192]
  __bf16* kcache   = qcache + (size_t)T * 3072;    // [T, 16, 192]
  __bf16* vcache   = kcache + (size_t)T * 3072;    // [T, 16, 128]
  __bf16* ob       = vcache + (size_t)T * 2048;    // [T, 16, 128]

  const dim3 blk(128);
  const int cgrid = 2048;

  // 0) bf16 conversions of GEMM operands
  f32_to_bf16_kernel<<<cgrid, 256, 0, stream>>>(hidden, hidden_b, (long)T * 2048);
  f32_to_bf16_kernel<<<cgrid, 256, 0, stream>>>(w_qkv_a, wqkv_b, 2048L * 2112);
  f32_to_bf16_kernel<<<cgrid, 256, 0, stream>>>(w_q_b, wqb_b, 1536L * 3072);
  f32_to_bf16_kernel<<<cgrid, 256, 0, stream>>>(w_kv_b, wkvb_b, 512L * 4096);
  f32_to_bf16_kernel<<<cgrid, 256, 0, stream>>>(w_o, wo_b, 2048L * 2048);

  // 1) qkv = hidden @ w_qkv_a
  gemm_bf16_kernel<<<dim3(2112 / 64, T / 128), blk, 0, stream>>>(
      hidden_b, wqkv_b, qkv, T, 2112, 2048);

  // 2) RMSNorm of q_c / kv_c slices -> bf16
  rmsnorm_kernel<<<T, 256, 0, stream>>>(qkv, 2112, 0, q_a_norm_w, qn, 1536);
  rmsnorm_kernel<<<T, 256, 0, stream>>>(qkv, 2112, 1536, kv_a_norm_w, kvn, 512);

  // 3) RoPE on k_pe slice -> kpe
  rope_k_kernel<<<T, 64, 0, stream>>>(qkv, positions, kpe, 2112, 2048);

  // 4) q = q_norm @ w_q_b, RoPE in-place per head
  gemm_bf16_kernel<<<dim3(3072 / 64, T / 128), blk, 0, stream>>>(
      qn, wqb_b, qb, T, 3072, 1536);
  rope_q_kernel<<<dim3(T, 16), 32, 0, stream>>>(qb, positions);

  // 5) kv = kv_norm @ w_kv_b
  gemm_bf16_kernel<<<dim3(4096 / 64, T / 128), blk, 0, stream>>>(
      kvn, wkvb_b, kvb, T, 4096, 512);

  // 6) bf16 attention caches, then flash attention
  build_caches_kernel<<<T, 256, 0, stream>>>(qb, kvb, kpe, qcache, kcache, vcache);
  mla_attn_kernel<<<dim3(T / 64, 16), blk, 0, stream>>>(qcache, kcache, vcache, ob, T);

  // 7) output projection
  gemm_bf16_kernel<<<dim3(2048 / 64, T / 128), blk, 0, stream>>>(
      ob, wo_b, out, T, 2048, 2048);
}
